// H_Attention_Graph_10230612099517
// MI455X (gfx1250) — compile-verified, matrix operand order checked by reading, not run on hardware
//
#include <hip/hip_runtime.h>

typedef __bf16 bf16;
typedef bf16  v16bf __attribute__((ext_vector_type(16)));
typedef float v8f   __attribute__((ext_vector_type(8)));
typedef unsigned u32x4 __attribute__((ext_vector_type(4)));
typedef unsigned u32x8 __attribute__((ext_vector_type(8)));

__device__ __forceinline__ bf16 f2bf(float f) {
  unsigned u = __builtin_bit_cast(unsigned, f);
  unsigned r = u + 0x7fffu + ((u >> 16) & 1u);   // round-to-nearest-even
  unsigned short h = (unsigned short)(r >> 16);
  return __builtin_bit_cast(bf16, h);
}

// ---------------------------------------------------------------------------
// conv1: x[512][3][112][112] f32, w[20][75] -> pooled bf16 [512][20][54][54]
// Implicit GEMM; M-rows arranged as 2x2 pool quads -> in-register maxpool+ReLU.
// Incremental im2col counters, guard only in the K-tail step.
// ---------------------------------------------------------------------------
template <bool TAIL>
__device__ __forceinline__ void conv1Step(const float* __restrict__ xin, int pixbase,
                                          const float* __restrict__ wrow, int kb,
                                          v8f& acc) {
  v16bf a, bm;
  int k = kb;
  int c = k / 25, rr = k - c * 25;
  int ky = rr / 5, kx = rr - ky * 5;
  int aoff = c * 12544 + ky * 112 + kx;
#pragma unroll
  for (int e = 0; e < 16; ++e) {
    float va = 0.f, vb = 0.f;
    if (!TAIL || k < 75) {
      va = xin[pixbase + aoff];
      vb = wrow[k];
    }
    a[e] = f2bf(va);
    bm[e] = f2bf(vb);
    ++k; ++kx; ++aoff;
    if (kx == 5) {
      kx = 0; ++ky; aoff += 107;              // +112-5
      if (ky == 5) { ky = 0; aoff += 11984; } // +12544-4*112-4-108
    }
  }
  acc = __builtin_amdgcn_wmma_f32_16x16x32_bf16(false, a, false, bm,
                                                (short)0, acc, false, false);
}

__global__ void __launch_bounds__(256) kConv1(const float* __restrict__ x,
                                              const float* __restrict__ w,
                                              const float* __restrict__ bias,
                                              bf16* __restrict__ out) {
  const int GROUPS = 729;                     // 2916 pooled pixels / 4
  int wid  = (blockIdx.x * blockDim.x + threadIdx.x) >> 5;
  int lane = threadIdx.x & 31;
  int ntile = wid & 1;
  int rest  = wid >> 1;
  int g = rest % GROUPS;
  int n = rest / GROUPS;
  if (n >= 512) return;                       // wave-uniform
  int half = lane >> 4, lidx = lane & 15;
  int p  = g * 4 + (lidx >> 2);               // pooled pixel for A-row lidx
  int oy = (p / 54) * 2 + ((lidx >> 1) & 1);
  int ox = (p % 54) * 2 + (lidx & 1);
  int pixbase = oy * 112 + ox;
  const float* xin = x + (size_t)n * 37632;   // 3*112*112
  int oc = ntile * 16 + lidx;
  const float* wrow = w + (size_t)(oc < 20 ? oc : 19) * 75;  // clamp, no guard
  v8f acc = {};
  conv1Step<false>(xin, pixbase, wrow, half * 16,       acc);
  conv1Step<false>(xin, pixbase, wrow, 32 + half * 16,  acc);
  conv1Step<true >(xin, pixbase, wrow, 64 + half * 16,  acc);
  if (oc < 20) {
    float bv = bias[oc];
    float m0 = fmaxf(fmaxf(acc[0], acc[1]), fmaxf(acc[2], acc[3]));
    float m1 = fmaxf(fmaxf(acc[4], acc[5]), fmaxf(acc[6], acc[7]));
    int p0 = g * 4 + half * 2;
    size_t ob = ((size_t)n * 20 + oc) * 2916;
    out[ob + p0]     = f2bf(fmaxf(m0 + bv, 0.f));
    out[ob + p0 + 1] = f2bf(fmaxf(m1 + bv, 0.f));
  }
}

// ---------------------------------------------------------------------------
// conv2: in bf16 [512][20][54][54], w[50][500] -> pooled bf16 [512][50][25][25]
// ---------------------------------------------------------------------------
template <bool TAIL>
__device__ __forceinline__ void conv2Step(const bf16* __restrict__ xi, int pixbase,
                                          const float* __restrict__ wrow, int kb,
                                          v8f& acc) {
  v16bf a, bm;
  int k = kb;
  int c = k / 25, rr = k - c * 25;
  int ky = rr / 5, kx = rr - ky * 5;
  int aoff = c * 2916 + ky * 54 + kx;
#pragma unroll
  for (int e = 0; e < 16; ++e) {
    bf16 va = f2bf(0.f);
    float vb = 0.f;
    if (!TAIL || k < 500) {
      va = xi[pixbase + aoff];
      vb = wrow[k];
    }
    a[e] = va;
    bm[e] = f2bf(vb);
    ++k; ++kx; ++aoff;
    if (kx == 5) {
      kx = 0; ++ky; aoff += 49;               // +54-5
      if (ky == 5) { ky = 0; aoff += 2646; }  // +2916-4*54-4-50
    }
  }
  acc = __builtin_amdgcn_wmma_f32_16x16x32_bf16(false, a, false, bm,
                                                (short)0, acc, false, false);
}

__global__ void __launch_bounds__(256) kConv2(const bf16* __restrict__ xin,
                                              const float* __restrict__ w,
                                              const float* __restrict__ bias,
                                              bf16* __restrict__ out) {
  const int GROUPS = 157;                     // ceil(625/4)
  int wid  = (blockIdx.x * blockDim.x + threadIdx.x) >> 5;
  int lane = threadIdx.x & 31;
  int ntile = wid & 3;
  int rest  = wid >> 2;
  int g = rest % GROUPS;
  int n = rest / GROUPS;
  if (n >= 512) return;
  int half = lane >> 4, lidx = lane & 15;
  int p = g * 4 + (lidx >> 2);
  int pp = (p < 625) ? p : 0;                 // clamp; bogus rows discarded on store
  int oy = (pp / 25) * 2 + ((lidx >> 1) & 1);
  int ox = (pp % 25) * 2 + (lidx & 1);
  int pixbase = oy * 54 + ox;
  const bf16* xi = xin + (size_t)n * 58320;   // 20*2916
  int oc = ntile * 16 + lidx;
  const float* wrow = w + (size_t)(oc < 50 ? oc : 49) * 500;
  v8f acc = {};
#pragma unroll
  for (int ks = 0; ks < 15; ++ks)
    conv2Step<false>(xi, pixbase, wrow, ks * 32 + half * 16, acc);
  conv2Step<true>(xi, pixbase, wrow, 480 + half * 16, acc);
  if (oc < 50) {
    float bv = bias[oc];
    float m0 = fmaxf(fmaxf(acc[0], acc[1]), fmaxf(acc[2], acc[3]));
    float m1 = fmaxf(fmaxf(acc[4], acc[5]), fmaxf(acc[6], acc[7]));
    int p0 = g * 4 + half * 2;
    size_t ob = (size_t)n * 31250 + (size_t)oc * 625;
    if (p0 < 625)     out[ob + p0]     = f2bf(fmaxf(m0 + bv, 0.f));
    if (p0 + 1 < 625) out[ob + p0 + 1] = f2bf(fmaxf(m1 + bv, 0.f));
  }
}

// ---------------------------------------------------------------------------
// fc: A bf16 [512][31250] x fc_w f32 [500][31250]^T -> H f32 [512][500], ReLU.
// One block per 16-channel N-tile; 8 waves x 4 M-tiles cover all 512 rows.
// B tile (16 rows x 32 k, f32) staged into LDS by the Tensor Data Mover
// (tensor_load_to_lds, TENSORcnt), double-buffered; TDM zero-pads the K tail
// and the rows >= 500 via the tensor_dim OOB rules.
// ---------------------------------------------------------------------------
__global__ void __launch_bounds__(256) kFC(const bf16* __restrict__ A,
                                           const float* __restrict__ w,
                                           const float* __restrict__ bias,
                                           float* __restrict__ H) {
  __shared__ float btile[2][16 * 32];         // LDS offsets 0 and 2048
  int nt   = blockIdx.x;                      // 0..31
  int wv   = threadIdx.x >> 5;
  int lane = threadIdx.x & 31;
  int half = lane >> 4, lidx = lane & 15;
  const int NSTEP = 977;                      // ceil(31250/32)

  unsigned remRows = (unsigned)(500 - nt * 16);               // 4..500
  unsigned long long wbase =
      (unsigned long long)(size_t)(w + (size_t)nt * 16 * 31250);

  const bf16* ar[4];
#pragma unroll
  for (int j = 0; j < 4; ++j) {
    int mt = wv * 4 + j;
    ar[j] = A + (size_t)(mt * 16 + lidx) * 31250;
  }
  v8f acc[4] = {};

  auto issueB = [&](int ks, int buf) {
    int k0 = ks * 32;
    unsigned long long ga = wbase + (unsigned long long)k0 * 4ull;
    unsigned remK = (unsigned)(31250 - k0);
    u32x4 g0;
    g0[0] = 1u;                                              // count=1, user mode
    g0[1] = (unsigned)(buf * 2048);                          // lds_addr
    g0[2] = (unsigned)(ga & 0xffffffffull);                  // global_addr lo
    g0[3] = (unsigned)((ga >> 32) & 0x01ffffffull) | (2u << 30);  // addr hi | type=2
    u32x8 g1;
    g1[0] = 0x00020000u;                 // workgroup_mask=0, data_size=2 (4B)
    g1[1] = remK << 16;                  // tensor_dim0[15:0] (remaining k)
    g1[2] = remRows << 16;               // dim0 hi=0 | tensor_dim1[15:0]
    g1[3] = 32u << 16;                   // dim1 hi=0 | tile_dim0=32
    g1[4] = 16u;                         // tile_dim1=16, tile_dim2=0
    g1[5] = 31250u;                      // tensor_dim0_stride lo
    g1[6] = 0u;                          // stride hi | dim1_stride lo
    g1[7] = 0u;
    asm volatile("tensor_load_to_lds %0, %1" :: "s"(g0), "s"(g1) : "memory");
  };

  if (wv == 0) issueB(0, 0);
  __builtin_amdgcn_s_wait_tensorcnt(0);
  __syncthreads();

  for (int ks = 0; ks < NSTEP; ++ks) {
    int buf = ks & 1;
    if (wv == 0 && ks + 1 < NSTEP) issueB(ks + 1, buf ^ 1);  // prefetch next tile
    int kb = ks * 32 + half * 16;
    const float* bl = &btile[buf][lidx * 32 + half * 16];
    v16bf bv;
#pragma unroll
    for (int e = 0; e < 16; ++e) bv[e] = f2bf(bl[e]);
    if (ks < NSTEP - 1) {                                    // unguarded main body
#pragma unroll
      for (int j = 0; j < 4; ++j) {
        const bf16* ap = ar[j] + kb;
        __builtin_prefetch(ap + 32, 0, 0);                   // global_prefetch_b8
        v16bf av;
#pragma unroll
        for (int e = 0; e < 16; ++e) av[e] = ap[e];
        acc[j] = __builtin_amdgcn_wmma_f32_16x16x32_bf16(false, av, false, bv,
                                                         (short)0, acc[j], false, false);
      }
    } else {                                                 // guarded K tail
#pragma unroll
      for (int j = 0; j < 4; ++j) {
        v16bf av;
#pragma unroll
        for (int e = 0; e < 16; ++e) {
          int k = kb + e;
          av[e] = (k < 31250) ? ar[j][k] : f2bf(0.f);
        }
        acc[j] = __builtin_amdgcn_wmma_f32_16x16x32_bf16(false, av, false, bv,
                                                         (short)0, acc[j], false, false);
      }
    }
    __builtin_amdgcn_s_wait_tensorcnt(0);
    __syncthreads();
  }

  int nc = nt * 16 + lidx;
  if (nc < 500) {
    float bb = bias[nc];
#pragma unroll
    for (int j = 0; j < 4; ++j) {
      int mt = wv * 4 + j;
#pragma unroll
      for (int r = 0; r < 8; ++r) {
        int row = mt * 16 + r + 8 * half;
        H[(size_t)row * 500 + nc] = fmaxf(acc[j][r] + bb, 0.f);
      }
    }
  }
}

// ---------------------------------------------------------------------------
// WMMA version of t = x @ w^T for dout=256 GCN layers.
// x: [8][kn][din] f32 compact, w: [256][din], t: [8][kn][256].
// Rows >= kn are clamped on load and masked on store.
// ---------------------------------------------------------------------------
__global__ void __launch_bounds__(256) kXWmma(const float* __restrict__ x,
                                              const float* __restrict__ w,
                                              float* __restrict__ t,
                                              int kn, int din, int mtiles) {
  int wid  = (blockIdx.x * blockDim.x + threadIdx.x) >> 5;
  int lane = threadIdx.x & 31;
  int nt = wid & 15;
  int rest = wid >> 4;
  int mt = rest % mtiles;
  int pa = rest / mtiles;
  if (pa >= 8) return;
  int half = lane >> 4, lidx = lane & 15;
  int row = mt * 16 + lidx;
  int rowc = (row < kn) ? row : (kn - 1);
  const float* xr = x + ((size_t)pa * kn + rowc) * din;
  const float* wr = w + (size_t)(nt * 16 + lidx) * din;
  v8f acc = {};
  for (int ks = 0; ks < din / 32; ++ks) {
    int kb = ks * 32 + half * 16;
    v16bf av, bv;
#pragma unroll
    for (int e = 0; e < 16; ++e) {
      av[e] = f2bf(xr[kb + e]);
      bv[e] = f2bf(wr[kb + e]);
    }
    acc = __builtin_amdgcn_wmma_f32_16x16x32_bf16(false, av, false, bv,
                                                  (short)0, acc, false, false);
  }
  int o = nt * 16 + lidx;
#pragma unroll
  for (int r = 0; r < 8; ++r) {
    int i = mt * 16 + r + 8 * half;
    if (i < kn) t[((size_t)pa * kn + i) * 256 + o] = acc[r];
  }
}

// ---------------------------------------------------------------------------
// Graph stage scalar kernels (tiny fraction of FLOPs).
// ---------------------------------------------------------------------------
__global__ void kFeat(const float* __restrict__ H, const float* __restrict__ info,
                      const float* __restrict__ lnf_g, const float* __restrict__ lnf_b,
                      const float* __restrict__ pos_w, const float* __restrict__ pos_b,
                      const float* __restrict__ lnp_g, const float* __restrict__ lnp_b,
                      float* __restrict__ feat) {
  int pa = blockIdx.x;
  int wave = threadIdx.x >> 5, lane = threadIdx.x & 31;
  for (int nd = wave; nd < 64; nd += 8) {
    int gidx = pa * 64 + nd;
    const float* h = H + (size_t)gidx * 500;
    float s = 0.f, ss = 0.f;
    for (int d = lane; d < 500; d += 32) { float v = h[d]; s += v; ss += v * v; }
    for (int msk = 16; msk > 0; msk >>= 1) {
      s  += __shfl_xor(s,  msk, 32);
      ss += __shfl_xor(ss, msk, 32);
    }
    float mu  = s / 500.f;
    float inv = rsqrtf(ss / 500.f - mu * mu + 1e-5f);
    float* F = feat + ((size_t)pa * 64 + nd) * 512;
    for (int d = lane; d < 500; d += 32)
      F[d] = (h[d] - mu) * inv * lnf_g[d] + lnf_b[d];
    const float* ii = info + (size_t)gidx * 6;
    float po[12]; float ps = 0.f, pss = 0.f;
#pragma unroll
    for (int j = 0; j < 12; ++j) {
      float v = pos_b[j];
#pragma unroll
      for (int d = 0; d < 6; ++d) v += ii[d] * pos_w[j * 6 + d];
      po[j] = v; ps += v; pss += v * v;
    }
    float pmu = ps / 12.f;
    float pinv = rsqrtf(pss / 12.f - pmu * pmu + 1e-5f);
    if (lane < 12) F[500 + lane] = (po[lane] - pmu) * pinv * lnp_g[lane] + lnp_b[lane];
  }
}

__global__ void kAdj(const float* __restrict__ feat, float* __restrict__ adj) {
  int pa = blockIdx.x;
  const float* F = feat + (size_t)pa * 64 * 512;
  __shared__ float sq[64];
  __shared__ float red[256];
  __shared__ float thr_s;
  for (int i = threadIdx.x; i < 64; i += blockDim.x) {
    const float* f = F + (size_t)i * 512;
    float s = 0.f;
    for (int d = 0; d < 512; ++d) s += f[d] * f[d];
    sq[i] = s;
  }
  __syncthreads();
  float* A = adj + (size_t)pa * 64 * 64;
  float lmax = -3.4e38f;
  for (int idx = threadIdx.x; idx < 64 * 64; idx += blockDim.x) {
    int i = idx >> 6, j = idx & 63;
    const float* fi = F + (size_t)i * 512;
    const float* fj = F + (size_t)j * 512;
    float dot = 0.f;
    for (int d = 0; d < 512; ++d) dot += fi[d] * fj[d];
    float dist = sq[i] + sq[j] - 2.f * dot;
    A[idx] = dist;
    lmax = fmaxf(lmax, dist);
  }
  red[threadIdx.x] = lmax;
  __syncthreads();
  for (int s = 128; s > 0; s >>= 1) {
    if ((int)threadIdx.x < s) red[threadIdx.x] = fmaxf(red[threadIdx.x], red[threadIdx.x + s]);
    __syncthreads();
  }
  if (threadIdx.x == 0) thr_s = 0.5f * red[0];   // T = 0.5
  __syncthreads();
  float thr = thr_s;
  for (int idx = threadIdx.x; idx < 64 * 64; idx += blockDim.x) {
    int i = idx >> 6, j = idx & 63;
    float dist = A[idx];                         // own element, no race
    A[idx] = (dist < thr && j > i) ? 1.f : 0.f;
  }
}

// scalar t = x @ w^T (used for the dout=1 SAGPool scorers)
__global__ void kXW(const float* __restrict__ x, const float* __restrict__ w,
                    float* __restrict__ t, int kn, int din, int dout) {
  int idx = blockIdx.x * blockDim.x + threadIdx.x;
  int total = 8 * kn * dout;
  if (idx >= total) return;
  int o = idx % dout; int rest = idx / dout;
  int i = rest % kn;  int pa = rest / kn;
  const float* xr = x + ((size_t)pa * kn + i) * din;
  const float* wr = w + (size_t)o * din;
  float s = 0.f;
  for (int d = 0; d < din; ++d) s += xr[d] * wr[d];
  t[idx] = s;
}

// out[j][o] = dinv[j] * sum_i (adj[i][j] + I) * dinv[i] * t[i][o] + b[o]
__global__ void kAgg(const float* __restrict__ adj, const float* __restrict__ t,
                     const float* __restrict__ bias, float* __restrict__ out,
                     int kn, int dout, int doRelu) {
  int pa = blockIdx.x;
  const float* A = adj + (size_t)pa * kn * kn;
  const float* T = t + (size_t)pa * kn * dout;
  float* O = out + (size_t)pa * kn * dout;
  __shared__ float dinv[64];
  for (int j = threadIdx.x; j < kn; j += blockDim.x) {
    float deg = 1.f;
    for (int i = 0; i < kn; ++i) deg += A[i * kn + j];
    dinv[j] = rsqrtf(deg);
  }
  __syncthreads();
  int total = kn * dout;
  for (int idx = threadIdx.x; idx < total; idx += blockDim.x) {
    int o = idx % dout, j = idx / dout;
    float s = 0.f;
    for (int i = 0; i < kn; ++i) {
      float a = A[i * kn + j] + ((i == j) ? 1.f : 0.f);
      s += a * dinv[i] * T[i * dout + o];
    }
    s = s * dinv[j] + bias[o];
    if (doRelu) s = fmaxf(s, 0.f);
    O[idx] = s;
  }
}

// SAGPool: rank-based top-k (matches lax.top_k tie-break), tanh gate, sub-adj.
__global__ void kPool(const float* __restrict__ x, const float* __restrict__ adj,
                      const float* __restrict__ score,
                      float* __restrict__ xo, float* __restrict__ adjo,
                      int kn, int kNew, int dout) {
  int pa = blockIdx.x;
  const float* S = score + (size_t)pa * kn;
  __shared__ float s[64];
  __shared__ int perm[64];
  for (int i = threadIdx.x; i < kn; i += blockDim.x) s[i] = S[i];
  __syncthreads();
  for (int i = threadIdx.x; i < kn; i += blockDim.x) {
    float si = s[i];
    int rank = 0;
    for (int j = 0; j < kn; ++j) {
      float sj = s[j];
      if (sj > si || (sj == si && j < i)) ++rank;
    }
    if (rank < kNew) perm[rank] = i;
  }
  __syncthreads();
  const float* X = x + (size_t)pa * kn * dout;
  float* XO = xo + (size_t)pa * kNew * dout;
  for (int idx = threadIdx.x; idx < kNew * dout; idx += blockDim.x) {
    int o = idx % dout, r = idx / dout;
    int src = perm[r];
    XO[idx] = X[src * dout + o] * tanhf(s[src]);
  }
  const float* AD = adj + (size_t)pa * kn * kn;
  float* AO = adjo + (size_t)pa * kNew * kNew;
  for (int idx = threadIdx.x; idx < kNew * kNew; idx += blockDim.x) {
    int c = idx % kNew, r = idx / kNew;
    AO[idx] = AD[perm[r] * kn + perm[c]];
  }
}

__global__ void kReadout(const float* __restrict__ z, float* __restrict__ inst,
                         int kn, int accumulate) {
  int pa = blockIdx.x;
  const float* Z = z + (size_t)pa * kn * 256;
  float* I = inst + (size_t)pa * 512;
  for (int o = threadIdx.x; o < 256; o += blockDim.x) {
    float mx = -3.4e38f, sm = 0.f;
    for (int i = 0; i < kn; ++i) { float v = Z[i * 256 + o]; mx = fmaxf(mx, v); sm += v; }
    float mean = sm / (float)kn;
    if (accumulate) { I[o] += mx; I[256 + o] += mean; }
    else            { I[o]  = mx; I[256 + o]  = mean; }
  }
}

__global__ void kAttn(const float* __restrict__ inst,
                      const float* __restrict__ w1, const float* __restrict__ b1,
                      const float* __restrict__ w2, const float* __restrict__ b2,
                      const float* __restrict__ cw, const float* __restrict__ cb,
                      const float* __restrict__ y, float* __restrict__ out) {
  __shared__ float hsh[1024];
  __shared__ float ash[8];
  __shared__ float aw[8];
  __shared__ float red[256];
  int t = threadIdx.x;
  for (int idx = t; idx < 1024; idx += 256) {
    int p = idx >> 7, j = idx & 127;
    const float* ir = inst + p * 512;
    const float* wr = w1 + j * 512;
    float s = b1[j];
    for (int d = 0; d < 512; ++d) s += ir[d] * wr[d];
    hsh[idx] = tanhf(s);
  }
  __syncthreads();
  if (t < 8) {
    float s = b2[0];
    for (int j = 0; j < 128; ++j) s += hsh[t * 128 + j] * w2[j];
    ash[t] = s;
  }
  __syncthreads();
  if (t == 0) {
    float mx = ash[0];
    for (int p = 1; p < 8; ++p) mx = fmaxf(mx, ash[p]);
    float sum = 0.f;
    for (int p = 0; p < 8; ++p) { float e = expf(ash[p] - mx); aw[p] = e; sum += e; }
    for (int p = 0; p < 8; ++p) aw[p] /= sum;
  }
  __syncthreads();
  float part = 0.f;
  for (int o = t; o < 512; o += 256) {
    float bg = 0.f;
    for (int p = 0; p < 8; ++p) bg += aw[p] * inst[p * 512 + o];
    part += bg * cw[o];
  }
  red[t] = part;
  __syncthreads();
  for (int s = 128; s > 0; s >>= 1) {
    if (t < s) red[t] += red[t + s];
    __syncthreads();
  }
  if (t == 0) {
    float logit = red[0] + cb[0];
    float prob = 1.f / (1.f + expf(-logit));
    float pred = (prob >= 0.5f) ? 1.f : 0.f;
    float pc = fminf(fmaxf(prob, 1e-5f), 1.f - 1e-5f);
    float yy = y[0];
    float loss = -(yy * logf(pc) + (1.f - yy) * logf(1.f - pc));
    out[0] = pc; out[1] = pred; out[2] = loss;
    for (int p = 0; p < 8; ++p) out[3 + p] = aw[p];
  }
}

// ---------------------------------------------------------------------------
extern "C" void kernel_launch(void* const* d_in, const int* in_sizes, int n_in,
                              void* d_out, int out_size, void* d_ws, size_t ws_size,
                              hipStream_t stream) {
  (void)in_sizes; (void)n_in; (void)out_size; (void)ws_size;
  const float* x       = (const float*)d_in[0];
  const float* info    = (const float*)d_in[1];
  const float* y       = (const float*)d_in[2];
  const float* conv1_w = (const float*)d_in[4];
  const float* conv1_b = (const float*)d_in[5];
  const float* conv2_w = (const float*)d_in[6];
  const float* conv2_b = (const float*)d_in[7];
  const float* fc_w    = (const float*)d_in[8];
  const float* fc_b    = (const float*)d_in[9];
  const float* lnf_g = (const float*)d_in[10]; const float* lnf_b = (const float*)d_in[11];
  const float* pos_w = (const float*)d_in[12]; const float* pos_b = (const float*)d_in[13];
  const float* lnp_g = (const float*)d_in[14]; const float* lnp_b = (const float*)d_in[15];
  const float* g1w = (const float*)d_in[16]; const float* g1b = (const float*)d_in[17];
  const float* g2w = (const float*)d_in[18]; const float* g2b = (const float*)d_in[19];
  const float* g3w = (const float*)d_in[20]; const float* g3b = (const float*)d_in[21];
  const float* p1w = (const float*)d_in[22]; const float* p1b = (const float*)d_in[23];
  const float* p2w = (const float*)d_in[24]; const float* p2b = (const float*)d_in[25];
  const float* p3w = (const float*)d_in[26]; const float* p3b = (const float*)d_in[27];
  const float* aw1 = (const float*)d_in[28]; const float* ab1 = (const float*)d_in[29];
  const float* aw2 = (const float*)d_in[30]; const float* ab2 = (const float*)d_in[31];
  const float* cw  = (const float*)d_in[32]; const float* cb  = (const float*)d_in[33];

  char* ws = (char*)d_ws;
  size_t off = 0;
  auto alloc = [&](size_t bytes) -> void* {
    void* p = ws + off;
    off = (off + bytes + 255) & ~(size_t)255;
    return p;
  };
  bf16*  in1p = (bf16*) alloc((size_t)512 * 20 * 2916 * 2);  // conv1 pooled bf16
  bf16*  h2   = (bf16*) alloc((size_t)512 * 31250 * 2);      // conv2 pooled bf16
  float* Hf   = (float*)alloc((size_t)512 * 500 * 4);        // fc out
  float* feat = (float*)alloc((size_t)8 * 64 * 512 * 4);
  float* adjA = (float*)alloc((size_t)8 * 64 * 64 * 4);
  float* adjB = (float*)alloc((size_t)8 * 64 * 64 * 4);
  float* zA   = (float*)alloc((size_t)8 * 64 * 256 * 4);
  float* zB   = (float*)alloc((size_t)8 * 64 * 256 * 4);
  float* tT   = (float*)alloc((size_t)8 * 64 * 256 * 4);
  float* sc   = (float*)alloc((size_t)8 * 64 * 4);
  float* inst = (float*)alloc((size_t)8 * 512 * 4);

  auto blocks = [](int total) { return (total + 255) / 256; };

  // CNN front-end (WMMA bf16 implicit GEMMs; FC stages B tiles via TDM)
  kConv1<<<93312, 256, 0, stream>>>(x, conv1_w, conv1_b, in1p);   // 512*729*2 waves
  kConv2<<<40192, 256, 0, stream>>>(in1p, conv2_w, conv2_b, h2);  // 512*157*4 waves
  kFC<<<32, 256, 0, stream>>>(h2, fc_w, fc_b, Hf);                // block per N-tile

  // Graph features + adjacency
  kFeat<<<8, 256, 0, stream>>>(Hf, info, lnf_g, lnf_b, pos_w, pos_b, lnp_g, lnp_b, feat);
  kAdj<<<8, 256, 0, stream>>>(feat, adjA);

  // layer 1 (kn=64, din=512) -> pool to 48
  kXWmma<<<64, 256, 0, stream>>>(feat, g1w, tT, 64, 512, 4);      // 8*4*16 waves
  kAgg<<<8, 256, 0, stream>>>(adjA, tT, g1b, zA, 64, 256, 1);
  kXW<<<blocks(8 * 64), 256, 0, stream>>>(zA, p1w, tT, 64, 256, 1);
  kAgg<<<8, 256, 0, stream>>>(adjA, tT, p1b, sc, 64, 1, 0);
  kPool<<<8, 256, 0, stream>>>(zA, adjA, sc, zB, adjB, 64, 48, 256);
  kReadout<<<8, 256, 0, stream>>>(zB, inst, 48, 0);

  // layer 2 (kn=48) -> pool to 36
  kXWmma<<<48, 256, 0, stream>>>(zB, g2w, tT, 48, 256, 3);        // 8*3*16 waves
  kAgg<<<8, 256, 0, stream>>>(adjB, tT, g2b, zA, 48, 256, 1);
  kXW<<<blocks(8 * 48), 256, 0, stream>>>(zA, p2w, tT, 48, 256, 1);
  kAgg<<<8, 256, 0, stream>>>(adjB, tT, p2b, sc, 48, 1, 0);
  kPool<<<8, 256, 0, stream>>>(zA, adjB, sc, zB, adjA, 48, 36, 256);
  kReadout<<<8, 256, 0, stream>>>(zB, inst, 36, 1);

  // layer 3 (kn=36) -> pool to 27
  kXWmma<<<48, 256, 0, stream>>>(zB, g3w, tT, 36, 256, 3);        // rows 36..47 masked
  kAgg<<<8, 256, 0, stream>>>(adjA, tT, g3b, zA, 36, 256, 1);
  kXW<<<blocks(8 * 36), 256, 0, stream>>>(zA, p3w, tT, 36, 256, 1);
  kAgg<<<8, 256, 0, stream>>>(adjA, tT, p3b, sc, 36, 1, 0);
  kPool<<<8, 256, 0, stream>>>(zA, adjA, sc, zB, adjB, 36, 27, 256);
  kReadout<<<8, 256, 0, stream>>>(zB, inst, 27, 1);

  // MIL attention + classifier + loss
  kAttn<<<1, 256, 0, stream>>>(inst, aw1, ab1, aw2, ab2, cw, cb, y, (float*)d_out);
}
